// RelativeBiasMQA_70849780515242
// MI455X (gfx1250) — matrix operand order repaired and checked by reading, not compiled, not run on hardware
//
#include <hip/hip_runtime.h>
#include <hip/hip_bf16.h>

#define S_LEN   2048
#define DMODEL  1024
#define NHEADS  16
#define DHEAD   64
#define BATCH   4

typedef __attribute__((ext_vector_type(16))) __bf16 v16bf;
typedef __attribute__((ext_vector_type(8)))  __bf16 v8bf;
typedef __attribute__((ext_vector_type(4)))  __bf16 v4bf;
typedef __attribute__((ext_vector_type(8)))  float  v8f;
typedef __attribute__((ext_vector_type(4)))  int    v4i;

#define WMMA_BF16(a, b, c) \
  __builtin_amdgcn_wmma_f32_16x16x32_bf16(false, (a), false, (b), (short)0, (c), false, false)

// Optional gfx1250 async global->LDS path (ASYNCcnt-tracked DMA).
#if defined(__gfx1250__) && __has_builtin(__builtin_amdgcn_global_load_async_to_lds_b128)
#define HAVE_ASYNC_LDS 1
typedef __attribute__((address_space(1))) v4i g_v4i;   // global-AS vector
typedef __attribute__((address_space(3))) v4i l_v4i;   // LDS-AS vector
#endif

// Optional VALU cross-lane permute (avoids ds_bpermute round-trips).
#if defined(__gfx1250__) && __has_builtin(__builtin_amdgcn_permlane16)
#define HAVE_PERMLANE16 1
#endif

__device__ __forceinline__ void wait_asynccnt0() {
#if defined(HAVE_ASYNC_LDS)
#if __has_builtin(__builtin_amdgcn_s_wait_asynccnt)
  __builtin_amdgcn_s_wait_asynccnt(0);
#else
  asm volatile("s_wait_asynccnt 0x0" ::: "memory");
#endif
#endif
}

// xor-shuffle within 16-lane groups (D in {1,2,4,8} keeps lanes in-group).
template <int D>
__device__ __forceinline__ float lane_xor16(float v) {
#if defined(HAVE_PERMLANE16)
  unsigned lo, hi;
  if      (D == 1) { lo = 0x67452301u; hi = 0xEFCDAB89u; }
  else if (D == 2) { lo = 0x54761032u; hi = 0xDCFE98BAu; }
  else if (D == 4) { lo = 0x32107654u; hi = 0xBA98FEDCu; }
  else             { lo = 0xFEDCBA98u; hi = 0x76543210u; }
  unsigned s = __float_as_uint(v);
  return __uint_as_float(__builtin_amdgcn_permlane16(s, s, lo, hi, false, false));
#else
  return __shfl_xor(v, D, 32);
#endif
}

// ---- WMMA fragment loaders (layouts per CDNA5 ISA 7.12.2, wave32) ----------

// A (16x32 bf16, MxK): lane l -> row m=l%16; elems 0..7 = K {kb..kb+7},
// elems 8..15 = K {16+kb..16+kb+7}, kb = (l<16)?0:8.  Two 16B loads.
__device__ __forceinline__ v16bf load_a_frag(const __bf16* row_base, int ld) {
  const int l  = threadIdx.x & 31;
  const int m  = l & 15;
  const int kb = (l < 16) ? 0 : 8;
  const __bf16* p = row_base + (size_t)m * ld;
  v8bf lo = *(const v8bf*)(p + kb);
  v8bf hi = *(const v8bf*)(p + 16 + kb);
  v16bf a;
#pragma unroll
  for (int i = 0; i < 8; ++i) { a[i] = lo[i]; a[8 + i] = hi[i]; }
  return a;
}

// B (32x16 bf16, KxN): lane l -> col n=l%16; elems 0..7 = K {kb..kb+7},
// elems 8..15 = K {16+kb..16+kb+7}, kb = (l<16)?0:8.
// col_base points at "row n holds K contiguous" storage (i.e. W[n][k] layout).
__device__ __forceinline__ v16bf load_b_frag(const __bf16* col_base, int ld) {
  const int l  = threadIdx.x & 31;
  const int n  = l & 15;
  const int kb = (l < 16) ? 0 : 8;
  const __bf16* p = col_base + (size_t)n * ld + kb;
  v8bf lo = *(const v8bf*)(p);
  v8bf hi = *(const v8bf*)(p + 16);
  v16bf b;
#pragma unroll
  for (int i = 0; i < 8; ++i) { b[i] = lo[i]; b[8 + i] = hi[i]; }
  return b;
}

// ---- fp32 -> bf16 conversion ------------------------------------------------
__global__ void cvt_kernel(const float* __restrict__ in, __bf16* __restrict__ out, int n4) {
  int i = blockIdx.x * 256 + threadIdx.x;
  if (i < n4) {
    float4 f = ((const float4*)in)[i];
    v4bf o;
    o[0] = (__bf16)f.x; o[1] = (__bf16)f.y; o[2] = (__bf16)f.z; o[3] = (__bf16)f.w;
    ((v4bf*)out)[i] = o;
  }
}

// ---- GEMM: out[M,N] = A[M,K] @ W[N,K]^T (+bias). 128 thr = 4 waves.
// Each wave computes a 16(M) x 16*NT(N) strip: the A fragment is loaded once
// per k-step and reused across NT WMMAs (2.5 x 16B loads per WMMA at NT=4). ---
template <int NT>
__global__ void __launch_bounds__(128)
gemm_bf16_kernel(const __bf16* __restrict__ A,
                 const __bf16* __restrict__ W,
                 __bf16* __restrict__ outB,
                 float*  __restrict__ outF,
                 const float* __restrict__ bias,
                 int M, int N, int K) {
  const int wave = threadIdx.x >> 5;
  const int mt = blockIdx.x * 16;
  const int nt = blockIdx.y * (64 * NT) + wave * (16 * NT);
  if (nt >= N) return;
  const __bf16* Ap = A + (size_t)mt * K;
  const __bf16* Wp = W + (size_t)nt * K;
  v8f c[NT];
#pragma unroll
  for (int t = 0; t < NT; ++t) c[t] = (v8f){};
#pragma unroll 2
  for (int k = 0; k < K; k += 32) {
    v16bf a = load_a_frag(Ap + k, K);
#pragma unroll
    for (int t = 0; t < NT; ++t) {
      v16bf b = load_b_frag(Wp + (size_t)t * 16 * K + k, K);
      c[t] = WMMA_BF16(a, b, c[t]);
    }
  }
  const int l  = threadIdx.x & 31;
  const int rb = (l < 16) ? 0 : 8;
#pragma unroll
  for (int t = 0; t < NT; ++t) {
    const int n = nt + t * 16 + (l & 15);
    const float bv = bias ? bias[n] : 0.0f;
#pragma unroll
    for (int r = 0; r < 8; ++r) {
      int m = mt + r + rb;
      float v = c[t][r] + bv;
      if (outF) outF[(size_t)m * N + n] = v;
      else      outB[(size_t)m * N + n] = (__bf16)v;
    }
  }
}

// ---- Flash attention: grid(S/64, H, B), 128 thr = 4 waves, wave = 16 q-rows.
// K-tile staged via async DMA; V-tile transposed (packed b32 stores); row max
// via permlane16 butterflies; row sum via an extra ones-WMMA. ----------------
__global__ void __launch_bounds__(128, 1)
attn_kernel(const __bf16* __restrict__ Q,
            const __bf16* __restrict__ Kt,
            const __bf16* __restrict__ Vt,
            const float* __restrict__ bias,
            __bf16* __restrict__ ctx) {
  const int b    = blockIdx.z;
  const int h    = blockIdx.y;
  const int tid  = threadIdx.x;
  const int wave = tid >> 5;
  const int q0   = blockIdx.x * 64 + wave * 16;

  __shared__ __align__(16) __bf16 Kls[32 * 64];      // [key][dh]
  __shared__ __align__(16) __bf16 Vls[64 * 40];      // [dh][key], padded stride 40
  __shared__ __align__(16) __bf16 Pls[4][16 * 32];   // per-wave P tile

  const __bf16* Qp = Q + ((size_t)b * S_LEN + q0) * DMODEL + h * DHEAD;
  const v16bf a0 = load_a_frag(Qp, DMODEL);        // dh 0..31
  const v16bf a1 = load_a_frag(Qp + 32, DMODEL);   // dh 32..63

  v16bf ones;                                      // all-ones B frag for rowsum
#pragma unroll
  for (int i = 0; i < 16; ++i) ones[i] = (__bf16)1.0f;

  const int l   = tid & 31;
  const int rb  = (l < 16) ? 0 : 8;
  const int col = l & 15;

  float mrow[8], lrow[8];
#pragma unroll
  for (int r = 0; r < 8; ++r) { mrow[r] = -1e30f; lrow[r] = 0.0f; }
  v8f o0 = {}, o1 = {}, o2 = {}, o3 = {};

  const float scale = 0.125f;  // 1/sqrt(64)
  const float* biasp = bias + (size_t)h * S_LEN * S_LEN + (size_t)q0 * S_LEN;

  for (int key0 = 0; key0 < S_LEN; key0 += 32) {
    __syncthreads();
    {   // cooperative stage: K tile linear (async), V tile transposed
      const uint4* Kg = (const uint4*)(Kt + ((size_t)b * S_LEN + key0) * DHEAD);
      uint4* Kd = (uint4*)Kls;
#if defined(HAVE_ASYNC_LDS)
      __builtin_amdgcn_global_load_async_to_lds_b128(
          (g_v4i*)(Kg + tid), (l_v4i*)(Kd + tid), 0, 0);
      __builtin_amdgcn_global_load_async_to_lds_b128(
          (g_v4i*)(Kg + tid + 128), (l_v4i*)(Kd + tid + 128), 0, 0);
#else
      Kd[tid]       = Kg[tid];
      Kd[tid + 128] = Kg[tid + 128];
#endif
      // V transpose: thread covers keys {key,key+1} x dh [dh0,dh0+8); two
      // adjacent keys pack into one 32-bit LDS store.
      const __bf16* Vrow = Vt + ((size_t)b * S_LEN + key0) * DHEAD;
      const int key = (tid & 15) * 2;
      const int dh0 = (tid >> 4) * 8;
      uint4 d0 = *(const uint4*)(Vrow + (size_t)key * DHEAD + dh0);
      uint4 d1 = *(const uint4*)(Vrow + (size_t)(key + 1) * DHEAD + dh0);
      const unsigned short* e0 = (const unsigned short*)&d0;
      const unsigned short* e1 = (const unsigned short*)&d1;
#pragma unroll
      for (int i = 0; i < 8; ++i) {
        unsigned pk = (unsigned)e0[i] | ((unsigned)e1[i] << 16);
        *(unsigned*)&Vls[(dh0 + i) * 40 + key] = pk;
      }
    }
    wait_asynccnt0();
    __syncthreads();

    // S = Q @ K^T : two 16x16 column tiles, K-dim = dh = 64 (2 WMMAs each)
    v8f s0 = {}, s1 = {};
    s0 = WMMA_BF16(a0, load_b_frag(Kls, 64), s0);
    s0 = WMMA_BF16(a1, load_b_frag(Kls + 32, 64), s0);
    s1 = WMMA_BF16(a0, load_b_frag(Kls + 16 * 64, 64), s1);
    s1 = WMMA_BF16(a1, load_b_frag(Kls + 16 * 64 + 32, 64), s1);

    // scale + bias; per-row local max
    float f0a[8], f1a[8], mx[8];
#pragma unroll
    for (int r = 0; r < 8; ++r) {
      const int row = r + rb;
      const float* bp = biasp + (size_t)row * S_LEN + key0;
      __builtin_prefetch(bp + 32, 0, 0);   // next k-step of the bias stream
      f0a[r] = s0[r] * scale + bp[col];
      f1a[r] = s1[r] * scale + bp[col + 16];
      mx[r]  = fmaxf(f0a[r], f1a[r]);
    }
    // batched xor-butterfly row-max over the 16 lanes sharing each row set
#pragma unroll
    for (int r = 0; r < 8; ++r) mx[r] = fmaxf(mx[r], lane_xor16<1>(mx[r]));
#pragma unroll
    for (int r = 0; r < 8; ++r) mx[r] = fmaxf(mx[r], lane_xor16<2>(mx[r]));
#pragma unroll
    for (int r = 0; r < 8; ++r) mx[r] = fmaxf(mx[r], lane_xor16<4>(mx[r]));
#pragma unroll
    for (int r = 0; r < 8; ++r) mx[r] = fmaxf(mx[r], lane_xor16<8>(mx[r]));

    __bf16* Pw = Pls[wave];
    float corr[8];
#pragma unroll
    for (int r = 0; r < 8; ++r) {
      const int row = r + rb;
      const float mnew = fmaxf(mrow[r], mx[r]);
      corr[r] = __expf(mrow[r] - mnew);
      mrow[r] = mnew;
      const float p0 = __expf(f0a[r] - mnew);
      const float p1 = __expf(f1a[r] - mnew);
      Pw[row * 32 + col]      = (__bf16)p0;
      Pw[row * 32 + col + 16] = (__bf16)p1;
      o0[r] *= corr[r]; o1[r] *= corr[r]; o2[r] *= corr[r]; o3[r] *= corr[r];
    }

    // P fragment; row sums via ones-WMMA (rowsum lands in every lane's c[r])
    const v16bf ap = load_a_frag(Pw, 32);
    v8f csum = WMMA_BF16(ap, ones, (v8f){});
#pragma unroll
    for (int r = 0; r < 8; ++r) lrow[r] = lrow[r] * corr[r] + csum[r];

    // O += P @ V  (K-dim = 32 keys, four 16-wide dh column tiles)
    o0 = WMMA_BF16(ap, load_b_frag(Vls + 0 * 16 * 40, 40), o0);
    o1 = WMMA_BF16(ap, load_b_frag(Vls + 1 * 16 * 40, 40), o1);
    o2 = WMMA_BF16(ap, load_b_frag(Vls + 2 * 16 * 40, 40), o2);
    o3 = WMMA_BF16(ap, load_b_frag(Vls + 3 * 16 * 40, 40), o3);
  }

  // normalize + write ctx in [b, s, h, dh] layout (bf16 feed for O-proj GEMM)
#pragma unroll
  for (int r = 0; r < 8; ++r) {
    const int q = q0 + r + rb;
    const float inv = 1.0f / lrow[r];
    const size_t base = (((size_t)b * S_LEN + q) * NHEADS + h) * DHEAD + col;
    ctx[base + 0]  = (__bf16)(o0[r] * inv);
    ctx[base + 16] = (__bf16)(o1[r] * inv);
    ctx[base + 32] = (__bf16)(o2[r] * inv);
    ctx[base + 48] = (__bf16)(o3[r] * inv);
  }
}

// ---- host orchestration -----------------------------------------------------
extern "C" void kernel_launch(void* const* d_in, const int* in_sizes, int n_in,
                              void* d_out, int out_size, void* d_ws, size_t ws_size,
                              hipStream_t stream) {
  (void)in_sizes; (void)n_in; (void)out_size; (void)ws_size;
  const float* x    = (const float*)d_in[0];
  const float* bias = (const float*)d_in[1];
  const float* w_q  = (const float*)d_in[2];
  const float* w_k  = (const float*)d_in[3];
  const float* w_v  = (const float*)d_in[4];
  const float* w_o  = (const float*)d_in[5];
  const float* b_o  = (const float*)d_in[6];
  float* out = (float*)d_out;

  char* ws = (char*)d_ws;
  size_t off = 0;
  auto carve = [&](size_t elems) { __bf16* p = (__bf16*)(ws + off); off += elems * 2; return p; };
  __bf16* x_bf  = carve((size_t)BATCH * S_LEN * DMODEL);  // 8192x1024
  __bf16* wq_bf = carve((size_t)DMODEL * DMODEL);
  __bf16* wk_bf = carve((size_t)DHEAD * DMODEL);
  __bf16* wv_bf = carve((size_t)DHEAD * DMODEL);
  __bf16* wo_bf = carve((size_t)DMODEL * DMODEL);
  __bf16* Q_bf  = carve((size_t)BATCH * S_LEN * DMODEL);
  __bf16* K_bf  = carve((size_t)BATCH * S_LEN * DHEAD);
  __bf16* V_bf  = carve((size_t)BATCH * S_LEN * DHEAD);
  __bf16* C_bf  = carve((size_t)BATCH * S_LEN * DMODEL);

  auto cvt = [&](const float* in, __bf16* o, int n) {
    int n4 = n / 4;
    cvt_kernel<<<(n4 + 255) / 256, 256, 0, stream>>>(in, o, n4);
  };
  cvt(x,   x_bf,  BATCH * S_LEN * DMODEL);
  cvt(w_q, wq_bf, DMODEL * DMODEL);
  cvt(w_k, wk_bf, DHEAD * DMODEL);
  cvt(w_v, wv_bf, DHEAD * DMODEL);
  cvt(w_o, wo_bf, DMODEL * DMODEL);

  const int M = BATCH * S_LEN;  // 8192
  // Q = x @ w_q^T  (bf16 out): waves cover 16x64 strips, block 16x256.
  gemm_bf16_kernel<4><<<dim3(M / 16, DMODEL / 256), 128, 0, stream>>>(
      x_bf, wq_bf, Q_bf, nullptr, nullptr, M, DMODEL, DMODEL);
  // K = x @ w_k^T, V = x @ w_v^T  (N = 64 -> NT=1, one block column)
  gemm_bf16_kernel<1><<<dim3(M / 16, 1), 128, 0, stream>>>(
      x_bf, wk_bf, K_bf, nullptr, nullptr, M, DHEAD, DMODEL);
  gemm_bf16_kernel<1><<<dim3(M / 16, 1), 128, 0, stream>>>(
      x_bf, wv_bf, V_bf, nullptr, nullptr, M, DHEAD, DMODEL);

  // flash attention with fused relative bias
  attn_kernel<<<dim3(S_LEN / 64, NHEADS, BATCH), 128, 0, stream>>>(
      Q_bf, K_bf, V_bf, bias, C_bf);

  // out = ctx @ w_o^T + b_o  (fp32 out)
  gemm_bf16_kernel<4><<<dim3(M / 16, DMODEL / 256), 128, 0, stream>>>(
      C_bf, wo_bf, nullptr, out, b_o, M, DMODEL, DMODEL);
}
